// GNN_80290118631700
// MI455X (gfx1250) — matrix operand order repaired and checked by reading, not compiled, not run on hardware
//
#include <hip/hip_runtime.h>
#include <hip/hip_bf16.h>

#define NN 50000
#define EE 800000
#define ET 850000          // E + N self loops
#define NH 4
#define CC 64
#define HC 256
#define FIN 128
#define EPB 16             // edges per block in accumulate pass

typedef __attribute__((ext_vector_type(16))) _Float16 v16h;
typedef __attribute__((ext_vector_type(8)))  float    v8f;

// ---------- helpers: order-preserving float<->uint for atomicMax ----------
__device__ __forceinline__ unsigned enc_f(float f) {
    unsigned u = __float_as_uint(f);
    return (u & 0x80000000u) ? ~u : (u | 0x80000000u);
}
__device__ __forceinline__ float dec_f(unsigned m) {
    unsigned u = (m & 0x80000000u) ? (m & 0x7FFFFFFFu) : ~m;
    return __uint_as_float(u);
}

// ---------- conversion kernels ----------
__global__ void gnn_f32_to_f16(const float* __restrict__ in, _Float16* __restrict__ out, int n) {
    int i = blockIdx.x * blockDim.x + threadIdx.x;
    if (i < n) out[i] = (_Float16)in[i];
}

// Wt[n*K + k] = W[k*Ncols + n]  (f32 -> f16 transpose)
__global__ void gnn_w_transpose_f16(const float* __restrict__ W, _Float16* __restrict__ Wt,
                                    int K, int Ncols) {
    int i = blockIdx.x * blockDim.x + threadIdx.x;
    if (i >= K * Ncols) return;
    int n = i / K, k = i % K;
    Wt[i] = (_Float16)W[(size_t)k * Ncols + n];
}

// ---------- WMMA GEMM: C[M,Ncols] = A[M,K](f16) * B[K,Ncols], B given as Bt[Ncols,K](f16)
__global__ void gnn_wmma_gemm(const _Float16* __restrict__ A, const _Float16* __restrict__ Bt,
                              float* __restrict__ C, int M, int K, int Ncols) {
    int gt   = blockIdx.x * blockDim.x + threadIdx.x;
    int wave = gt >> 5;
    int lane = threadIdx.x & 31;
    int ntiles = Ncols >> 4;
    int mt = wave / ntiles;
    int nt = wave % ntiles;
    if (mt * 16 >= M) return;                 // wave-uniform guard (EXEC stays all-1s)
    int kg = lane >> 4;                       // lane group 0/1
    int ml = lane & 15;
    const _Float16* arow = A  + (size_t)(mt * 16 + ml) * K;
    const _Float16* brow = Bt + (size_t)(nt * 16 + ml) * K;
    v8f acc = {};
    for (int k0 = 0; k0 < K; k0 += 32) {
        // A 16x32 fragment: lane holds A[m, k0+kg*8 .. +7] and A[m, k0+16+kg*8 .. +7]
        union { v16h v; int4 q[2]; } ua, ub;
        ua.q[0] = *reinterpret_cast<const int4*>(arow + k0 + kg * 8);
        ua.q[1] = *reinterpret_cast<const int4*>(arow + k0 + kg * 8 + 16);
        // B 32x16 fragment: lane holds column n, K = k0+kg*8 .. +7 and k0+16+kg*8 .. +7 (from Bt rows)
        ub.q[0] = *reinterpret_cast<const int4*>(brow + k0 + kg * 8);
        ub.q[1] = *reinterpret_cast<const int4*>(brow + k0 + kg * 8 + 16);
        acc = __builtin_amdgcn_wmma_f32_16x16x32_f16(false, ua.v, false, ub.v,
                                                     (short)0, acc, false, false);
    }
    // C/D layout: VGPR r -> row (kg*8 + r), col = ml
    int row0 = mt * 16 + kg * 8;
    int col  = nt * 16 + ml;
    for (int r = 0; r < 8; ++r)
        C[(size_t)(row0 + r) * Ncols + col] = acc[r];
}

// ---------- per-(node,head) attention dot products ----------
__global__ void gnn_attn_dots(const float* __restrict__ hfeat,
                              const float* __restrict__ a_src, const float* __restrict__ a_dst,
                              float* __restrict__ asrcN, float* __restrict__ adstN) {
    int idx = blockIdx.x * blockDim.x + threadIdx.x;
    if (idx >= NN * NH) return;
    int n = idx >> 2, hh = idx & 3;
    const float* hp = hfeat + (size_t)n * HC + hh * CC;
    const float* as = a_src + hh * CC;
    const float* ad = a_dst + hh * CC;
    float s1 = 0.f, s2 = 0.f;
    for (int c = 0; c < CC; ++c) { float v = hp[c]; s1 += v * as[c]; s2 += v * ad[c]; }
    asrcN[idx] = s1;
    adstN[idx] = s2;
}

// ---------- init: agg=0, denom=0, emax=enc(-inf) ----------
__global__ void gnn_init(unsigned* __restrict__ emax, float* __restrict__ denom,
                         float* __restrict__ agg) {
    int i = blockIdx.x * blockDim.x + threadIdx.x;
    if (i < NN * HC) agg[i] = 0.f;
    if (i < NN * NH) { emax[i] = 0x007FFFFFu; denom[i] = 0.f; }   // enc(-inf) = ~0xFF800000
}

// ---------- edge pass 1: e = leaky_relu(asrc[src]+adst[dst]); segment max ----------
__global__ void gnn_edge_max(const int* __restrict__ ei,
                             const float* __restrict__ asrcN, const float* __restrict__ adstN,
                             float* __restrict__ eedge, unsigned* __restrict__ emax) {
    int idx = blockIdx.x * blockDim.x + threadIdx.x;
    if (idx >= ET * NH) return;
    int e = idx >> 2, hh = idx & 3;
    int s, d;
    if (e < EE) { s = ei[e]; d = ei[EE + e]; } else { s = d = e - EE; }
    float v = asrcN[s * NH + hh] + adstN[d * NH + hh];
    v = v > 0.f ? v : 0.2f * v;
    eedge[idx] = v;
    atomicMax(&emax[d * NH + hh], enc_f(v));
}

// ---------- edge pass 2: w = exp(e - emax[dst]); denom += w; agg[dst] += w*h[src] ----------
__global__ void gnn_edge_accum(const int* __restrict__ ei, const float* __restrict__ eedge,
                               const unsigned* __restrict__ emax,
                               const float* __restrict__ hfeat,
                               float* __restrict__ denom, float* __restrict__ agg) {
    int t  = threadIdx.x;        // 256 threads = one channel each
    int hh = t >> 6;             // head = channel / 64
    int ebase = blockIdx.x * EPB;
    for (int k = 0; k < EPB; ++k) {
        int e = ebase + k;
        if (e >= ET) return;
        int s, d;
        if (e < EE) { s = ei[e]; d = ei[EE + e]; } else { s = d = e - EE; }
        // prefetch next source feature row (emits global_prefetch_b8)
        if (k + 1 < EPB && e + 1 < ET) {
            int sn = (e + 1 < EE) ? ei[e + 1] : (e + 1 - EE);
            __builtin_prefetch(hfeat + (size_t)sn * HC + t, 0, 0);
        }
        float mx = dec_f(emax[d * NH + hh]);
        float w  = expf(eedge[e * NH + hh] - mx);
        if ((t & 63) == 0) atomicAdd(&denom[d * NH + hh], w);
        float val = w * hfeat[(size_t)s * HC + t];
        atomicAdd(&agg[(size_t)d * HC + t], val);
    }
}

// ---------- node epilogue: normalize + bias + BN + ReLU ----------
__global__ void gnn_post_node(const float* __restrict__ agg, const float* __restrict__ denom,
                              const float* __restrict__ bias, const float* __restrict__ g,
                              const float* __restrict__ be, const float* __restrict__ mean,
                              const float* __restrict__ var,
                              _Float16* __restrict__ out16, float* __restrict__ out32) {
    int i = blockIdx.x * blockDim.x + threadIdx.x;
    if (i >= NN * HC) return;
    int n = i / HC, t = i % HC;
    float d   = denom[n * NH + (t >> 6)] + 1e-16f;
    float val = agg[i] / d + bias[t];
    val = (val - mean[t]) * rsqrtf(var[t] + 1e-5f) * g[t] + be[t];
    val = fmaxf(val, 0.f);
    if (out16) out16[i] = (_Float16)val;
    if (out32) out32[i] = val;
}

// ---------- log_softmax over 256 dims, one wave32 per node ----------
__global__ void gnn_log_softmax(const float* __restrict__ in, float* __restrict__ out) {
    int gt   = blockIdx.x * blockDim.x + threadIdx.x;
    int wid  = gt >> 5;
    int lane = threadIdx.x & 31;
    if (wid >= NN) return;
    const float* row = in + (size_t)wid * HC;
    float v[8];
    float mx = -INFINITY;
    for (int i = 0; i < 8; ++i) { v[i] = row[lane + 32 * i]; mx = fmaxf(mx, v[i]); }
    for (int o = 16; o >= 1; o >>= 1) mx = fmaxf(mx, __shfl_xor(mx, o, 32));
    float s = 0.f;
    for (int i = 0; i < 8; ++i) s += expf(v[i] - mx);
    for (int o = 16; o >= 1; o >>= 1) s += __shfl_xor(s, o, 32);
    float ls = logf(s);
    float* orow = out + (size_t)wid * HC;
    for (int i = 0; i < 8; ++i) orow[lane + 32 * i] = v[i] - mx - ls;
}

// =====================================================================
extern "C" void kernel_launch(void* const* d_in, const int* in_sizes, int n_in,
                              void* d_out, int out_size, void* d_ws, size_t ws_size,
                              hipStream_t stream) {
    const float* x      = (const float*)d_in[0];
    const int*   ei     = (const int*)d_in[1];
    const float* W1     = (const float*)d_in[2];
    const float* a_src1 = (const float*)d_in[3];
    const float* a_dst1 = (const float*)d_in[4];
    const float* b1     = (const float*)d_in[5];
    const float* g1     = (const float*)d_in[6];
    const float* be1    = (const float*)d_in[7];
    const float* m1     = (const float*)d_in[8];
    const float* v1     = (const float*)d_in[9];
    const float* W2     = (const float*)d_in[10];
    const float* a_src2 = (const float*)d_in[11];
    const float* a_dst2 = (const float*)d_in[12];
    const float* b2     = (const float*)d_in[13];
    const float* g2     = (const float*)d_in[14];
    const float* be2    = (const float*)d_in[15];
    const float* m2     = (const float*)d_in[16];
    const float* v2     = (const float*)d_in[17];
    float* outp = (float*)d_out;

    // ---- workspace carve-up (256B aligned) ----
    char* base = (char*)d_ws;
    size_t off = 0;
    auto carve = [&](size_t bytes) { char* p = base + off; off = (off + bytes + 255) & ~(size_t)255; return p; };
    float*     hbuf  = (float*)    carve((size_t)NN * HC * 4);   // GEMM output / gather source
    float*     agg   = (float*)    carve((size_t)NN * HC * 4);   // segment-sum accumulator
    _Float16*  a16   = (_Float16*) carve((size_t)NN * HC * 2);   // f16 GEMM A (layer1 uses NN*FIN)
    _Float16*  wt16  = (_Float16*) carve((size_t)HC * HC * 2);   // f16 transposed weights
    float*     asrcN = (float*)    carve((size_t)NN * NH * 4);
    float*     adstN = (float*)    carve((size_t)NN * NH * 4);
    unsigned*  emax  = (unsigned*) carve((size_t)NN * NH * 4);
    float*     denom = (float*)    carve((size_t)NN * NH * 4);
    float*     eedge = (float*)    carve((size_t)ET * NH * 4);
    (void)ws_size; (void)in_sizes; (void)n_in; (void)out_size;

    const int T = 256;
    auto cdiv = [](long long a, long long b) { return (int)((a + b - 1) / b); };

    // ================= Layer 1 =================
    gnn_f32_to_f16<<<cdiv((long long)NN * FIN, T), T, 0, stream>>>(x, a16, NN * FIN);
    gnn_w_transpose_f16<<<cdiv((long long)FIN * HC, T), T, 0, stream>>>(W1, wt16, FIN, HC);
    gnn_wmma_gemm<<<(NN / 16) * (HC / 16) * 32 / T, T, 0, stream>>>(a16, wt16, hbuf, NN, FIN, HC);
    gnn_attn_dots<<<cdiv((long long)NN * NH, T), T, 0, stream>>>(hbuf, a_src1, a_dst1, asrcN, adstN);
    gnn_init<<<cdiv((long long)NN * HC, T), T, 0, stream>>>(emax, denom, agg);
    gnn_edge_max<<<cdiv((long long)ET * NH, T), T, 0, stream>>>(ei, asrcN, adstN, eedge, emax);
    gnn_edge_accum<<<cdiv((long long)ET, EPB), T, 0, stream>>>(ei, eedge, emax, hbuf, denom, agg);
    gnn_post_node<<<cdiv((long long)NN * HC, T), T, 0, stream>>>(agg, denom, b1, g1, be1, m1, v1,
                                                                 a16, (float*)nullptr);

    // ================= Layer 2 =================
    gnn_w_transpose_f16<<<cdiv((long long)HC * HC, T), T, 0, stream>>>(W2, wt16, HC, HC);
    gnn_wmma_gemm<<<(NN / 16) * (HC / 16) * 32 / T, T, 0, stream>>>(a16, wt16, hbuf, NN, HC, HC);
    gnn_attn_dots<<<cdiv((long long)NN * NH, T), T, 0, stream>>>(hbuf, a_src2, a_dst2, asrcN, adstN);
    gnn_init<<<cdiv((long long)NN * HC, T), T, 0, stream>>>(emax, denom, agg);
    gnn_edge_max<<<cdiv((long long)ET * NH, T), T, 0, stream>>>(ei, asrcN, adstN, eedge, emax);
    gnn_edge_accum<<<cdiv((long long)ET, EPB), T, 0, stream>>>(ei, eedge, emax, hbuf, denom, agg);
    // epilogue writes f32 back into hbuf (gather source no longer needed)
    gnn_post_node<<<cdiv((long long)NN * HC, T), T, 0, stream>>>(agg, denom, b2, g2, be2, m2, v2,
                                                                 (_Float16*)nullptr, hbuf);
    gnn_log_softmax<<<cdiv((long long)NN * 32, T), T, 0, stream>>>(hbuf, outp);
}
